// BERTVergaPytorch_50096498540978
// MI455X (gfx1250) — matrix operand order repaired and checked by reading, not compiled, not run on hardware
//
#include <hip/hip_runtime.h>
#include <hip/hip_bf16.h>

#define S_ 500
#define D_ 768
#define K_ 16
#define R_ 256
#define P_ 64

typedef __attribute__((ext_vector_type(16))) __bf16 v16bf;
typedef __attribute__((ext_vector_type(8)))  float  v8f;

union Frag {
    v16bf bf;
    uint4 q[2];
};

// fp32 -> bf16 conversion of the word-embedding table
__global__ void cvt_emb_kernel(const float* __restrict__ we, __bf16* __restrict__ embT, int n) {
    int i = blockIdx.x * blockDim.x + threadIdx.x;
    if (i < n) embT[i] = (__bf16)we[i];
}

// W[d][k][e] (fp32) -> Wt[k][e][d] (bf16): stage-1 B fragments become contiguous along d
__global__ void cvt_w_kernel(const float* __restrict__ W, __bf16* __restrict__ Wt) {
    long long i = (long long)blockIdx.x * blockDim.x + threadIdx.x;
    const long long n = (long long)K_ * D_ * D_;
    if (i < n) {
        int d = (int)(i % D_);
        long long rem = i / D_;
        int e = (int)(rem % D_);
        int k = (int)(rem / D_);
        Wt[i] = (__bf16)W[((long long)d * K_ + k) * D_ + e];
    }
}

// One workgroup per (r, k): z = (E1_r @ W_k) @ E2_r^T  (64x64), then masked logsumexp.
__global__ __launch_bounds__(512, 1)
void biaffine_lse_kernel(const __bf16* __restrict__ embT, const __bf16* __restrict__ Wt,
                         const int* __restrict__ idx1, const int* __restrict__ idx2,
                         const float* __restrict__ mask1, const float* __restrict__ mask2,
                         float* __restrict__ out)
{
    const int r    = blockIdx.y;    // 0..255
    const int k    = blockIdx.x;    // 0..15
    const int tid  = threadIdx.x;   // 0..511 (16 wave32s)
    const int wave = tid >> 5;
    const int lane = tid & 31;
    const int half = lane >> 4;     // which 16-lane half
    const int l15  = lane & 15;
    const int pi   = wave >> 2;     // row-tile index (0..3)
    const int ci   = wave & 3;      // col-tile index: e-tile (stage1) / q-tile (stage2)

    __shared__ __bf16 e1s[P_][D_ + 8];  // gathered E1 rows, padded (388 dwords/row: conflict-free)
    __shared__ __bf16 hs[P_][P_ + 8];   // H e-tile (bf16), padded (36 dwords/row)
    __shared__ float  m1s[P_];
    __shared__ float  m2s[P_];
    __shared__ int    q2s[P_];
    __shared__ float  red[16];

    // ---- gather E1 rows straight into LDS via async-to-LDS (8 threads per row) ----
    {
        int j = tid >> 3, sub = tid & 7;
        int row = idx1[r * P_ + j];
        unsigned ldsa = (unsigned)(size_t)(&e1s[j][0]) + 16u * (unsigned)sub;
        unsigned long long ga =
            (unsigned long long)(const void*)(embT + (size_t)row * D_) + 16u * (unsigned)sub;
        #pragma unroll
        for (int i = 0; i < 12; ++i)
            asm volatile("global_load_async_to_lds_b128 %0, %1, off offset:%c2"
                         :: "v"(ldsa), "v"(ga), "i"(i * 128) : "memory");
    }
    if (tid < P_)            m1s[tid]          = mask1[r * P_ + tid];
    else if (tid < 2 * P_)   m2s[tid - P_]     = mask2[r * P_ + tid - P_];
    else if (tid < 3 * P_)   q2s[tid - 2 * P_] = idx2[r * P_ + tid - 2 * P_];
    asm volatile("s_wait_asynccnt 0x0" ::: "memory");
    __syncthreads();

    v8f zacc = {};
    const __bf16* aRow = &e1s[16 * pi + l15][0];   // A: row p = 16*pi + (lane&15)
    const __bf16* hRow = &hs[16 * pi + l15][0];

    for (int t = 0; t < 12; ++t) {
        // ---- stage 1: H tile (pi, ci) = E1[16pi.., :] @ Wt_k[:, 64t+16ci..]^T over d=768 ----
        v8f h = {};
        // B lane: column e = 64t + 16*ci + l15; elements = d chunks at 8*half and 16+8*half
        const __bf16* wB = Wt + ((size_t)k * D_ + (size_t)(64 * t + 16 * ci + l15)) * (size_t)D_
                              + 8 * half;
        const __bf16* aP = aRow + 8 * half;        // A: K chunks at 8*half and 16+8*half
        #pragma unroll
        for (int dk = 0; dk < 24; ++dk) {
            Frag a, b;
            const uint4* pa = (const uint4*)(aP + 32 * dk);
            a.q[0] = pa[0];          // K = 32dk + 8*half + 0..7
            a.q[1] = pa[2];          // K = 32dk + 16 + 8*half + 0..7
            const uint4* pb = (const uint4*)(wB + 32 * dk);
            __builtin_prefetch(wB + 32 * dk + 256, 0, 1);
            b.q[0] = pb[0];          // K = 32dk + 8*half + 0..7
            b.q[1] = pb[2];          // K = 32dk + 16 + 8*half + 0..7
            h = __builtin_amdgcn_wmma_f32_16x16x32_bf16(false, a.bf, false, b.bf,
                                                        (short)0, h, false, false);
        }

        // ---- spill H tile to LDS as bf16 (D layout: row = v + 8*half, col = l15) ----
        __syncthreads();   // previous iteration's stage-2 reads of hs are done
        #pragma unroll
        for (int v = 0; v < 8; ++v)
            hs[16 * pi + 8 * half + v][16 * ci + l15] = (__bf16)h[v];
        __syncthreads();

        // ---- stage 2: z tile (pi, ci) += H[16pi.., :] @ E2[16ci.., :]^T over e-tile (64) ----
        const int grow = q2s[16 * ci + l15];                        // gathered q row
        const __bf16* e2p = embT + (size_t)grow * D_ + 64 * t + 8 * half;
        const __bf16* hP  = hRow + 8 * half;
        #pragma unroll
        for (int ek = 0; ek < 2; ++ek) {
            Frag a, b;
            const uint4* pa = (const uint4*)(hP + 32 * ek);
            a.q[0] = pa[0];
            a.q[1] = pa[2];
            const uint4* pb = (const uint4*)(e2p + 32 * ek);
            b.q[0] = pb[0];
            b.q[1] = pb[2];
            zacc = __builtin_amdgcn_wmma_f32_16x16x32_bf16(false, a.bf, false, b.bf,
                                                           (short)0, zacc, false, false);
        }
        // barrier before next iteration's hs overwrite is the __syncthreads() above
    }

    // ---- masked logsumexp over the workgroup's 64x64 z ----
    const float mq = m2s[16 * ci + l15];
    float mloc = -3.0e38f;
    #pragma unroll
    for (int v = 0; v < 8; ++v) {
        float b = m1s[16 * pi + 8 * half + v] * mq;
        if (b > 0.0f) mloc = fmaxf(mloc, zacc[v]);
    }
    // wave32 reduction, then 16-entry LDS combine
    #pragma unroll
    for (int off = 16; off > 0; off >>= 1) mloc = fmaxf(mloc, __shfl_xor(mloc, off, 32));
    if (lane == 0) red[wave] = mloc;
    __syncthreads();
    float M = red[0];
    #pragma unroll
    for (int w = 1; w < 16; ++w) M = fmaxf(M, red[w]);
    __syncthreads();

    float sloc = 0.0f;
    #pragma unroll
    for (int v = 0; v < 8; ++v) {
        float b = m1s[16 * pi + 8 * half + v] * mq;
        if (b > 0.0f) sloc += __expf(zacc[v] - M);
    }
    #pragma unroll
    for (int off = 16; off > 0; off >>= 1) sloc += __shfl_xor(sloc, off, 32);
    if (lane == 0) red[wave] = sloc;
    __syncthreads();
    if (tid == 0) {
        float ssum = 0.0f;
        #pragma unroll
        for (int w = 0; w < 16; ++w) ssum += red[w];
        out[r * K_ + k] = M + __logf(ssum);
    }
}

extern "C" void kernel_launch(void* const* d_in, const int* in_sizes, int n_in,
                              void* d_out, int out_size, void* d_ws, size_t ws_size,
                              hipStream_t stream) {
    const float* we    = (const float*)d_in[0];
    const float* W     = (const float*)d_in[1];
    const int*   idx1  = (const int*)d_in[2];
    const int*   idx2  = (const int*)d_in[3];
    const float* mask1 = (const float*)d_in[4];
    const float* mask2 = (const float*)d_in[5];
    float* out = (float*)d_out;

    __bf16* embT = (__bf16*)d_ws;
    size_t embBytes = ((size_t)S_ * D_ * 2 + 255) / 256 * 256;
    __bf16* Wt = (__bf16*)((char*)d_ws + embBytes);

    int nE = S_ * D_;
    cvt_emb_kernel<<<(nE + 255) / 256, 256, 0, stream>>>(we, embT, nE);

    long long nW = (long long)K_ * D_ * D_;
    cvt_w_kernel<<<(int)((nW + 255) / 256), 256, 0, stream>>>(W, Wt);

    dim3 grid(K_, R_);
    biaffine_lse_kernel<<<grid, 512, 0, stream>>>(embT, Wt, idx1, idx2, mask1, mask2, out);
}
